// StudentModel_79259326480956
// MI455X (gfx1250) — matrix operand order repaired and checked by reading, not compiled, not run on hardware
//
#include <hip/hip_runtime.h>

#define B_ 4
#define N_ 96
#define H_ 768
#define H2_ 384
#define P_ 128
#define HEADS_ 12
#define D_ 64
#define LAYERS_ 3
#define REL_ 17

typedef __attribute__((ext_vector_type(16))) __bf16 v16bf;
typedef __attribute__((ext_vector_type(8)))  float  v8f;
typedef __attribute__((ext_vector_type(4)))  float  f32x4;
typedef __attribute__((ext_vector_type(2)))  unsigned int u32x2;

union FragBF { v16bf v; f32x4 f4[2]; };
union Pack4  { __bf16 h[4]; u32x2 u; };

__device__ __forceinline__ float sigmoidf_(float x) { return 1.f / (1.f + __expf(-x)); }

// ---------------------------------------------------------------------------
// bf16 WMMA GEMM: C(MxN) = act(A(MxK) @ B(KxN) + bias)
// REQUIRES: M % 64 == 0, N % 64 == 0, K % 64 == 0 (true for every call site).
// 64x64 C tile / block, 4 waves, BK=64 (2 wmma K-steps per LDS stage).
// ---------------------------------------------------------------------------
__global__ __launch_bounds__(128)
void k_gemm(const float* __restrict__ A, const float* __restrict__ Bm,
            const float* __restrict__ bias, float* __restrict__ C,
            int M, int N, int K, int lda, int ldb, int ldc, int act)
{
  __shared__ __align__(16) __bf16 As[64 * 64];   // [m][k]
  __shared__ __align__(16) __bf16 Bs[64 * 64];   // transposed: [n][k]
  const int bn0 = blockIdx.x * 64;
  const int bm0 = blockIdx.y * 64;
  const int tid = threadIdx.x;
  const int wave = tid >> 5;
  const int lane = tid & 31;
  const int hf   = lane >> 4;   // lane half (0: lanes 0-15, 1: lanes 16-31)
  const int l16  = lane & 15;

  v8f acc[4];
  for (int nt = 0; nt < 4; ++nt)
    for (int r = 0; r < 8; ++r) acc[nt][r] = 0.f;

  const int r_a  = (tid >> 4);        // A: thread's base row (row r_a + 0.. via idx4)
  const int nK = K >> 6;
  for (int kt = 0; kt < nK; ++kt) {
    const int bk0 = kt << 6;
    // bulk-load tiles into registers (b128, no guards, no per-load waits)
    f32x4 av[8], bv[8];
#pragma unroll
    for (int i = 0; i < 8; ++i) {
      int idx4 = tid + i * 128;           // 1024 float4 = 64x64 floats
      int r = idx4 >> 4, c4 = idx4 & 15;  // 16 float4 per 64-float row
      av[i] = *(const f32x4*)(A + (size_t)(bm0 + r) * lda + bk0 + c4 * 4);
    }
#pragma unroll
    for (int i = 0; i < 8; ++i) {
      int idx4 = tid + i * 128;
      int kk = idx4 >> 4, n4 = idx4 & 15;
      bv[i] = *(const f32x4*)(Bm + (size_t)(bk0 + kk) * ldb + bn0 + n4 * 4);
    }
    __syncthreads();   // previous stage's LDS readers are done
#pragma unroll
    for (int i = 0; i < 8; ++i) {
      int idx4 = tid + i * 128;
      int r = idx4 >> 4, c4 = idx4 & 15;
      Pack4 p;
      p.h[0] = (__bf16)av[i][0]; p.h[1] = (__bf16)av[i][1];
      p.h[2] = (__bf16)av[i][2]; p.h[3] = (__bf16)av[i][3];
      *(u32x2*)(As + r * 64 + c4 * 4) = p.u;
    }
#pragma unroll
    for (int i = 0; i < 8; ++i) {
      int idx4 = tid + i * 128;
      int kk = idx4 >> 4, n4 = idx4 & 15;
      Bs[(n4 * 4 + 0) * 64 + kk] = (__bf16)bv[i][0];
      Bs[(n4 * 4 + 1) * 64 + kk] = (__bf16)bv[i][1];
      Bs[(n4 * 4 + 2) * 64 + kk] = (__bf16)bv[i][2];
      Bs[(n4 * 4 + 3) * 64 + kk] = (__bf16)bv[i][3];
    }
    __syncthreads();
    if (kt + 1 < nK)   // speculative prefetch of next A tile (global_prefetch_b8)
      __builtin_prefetch(A + (size_t)(bm0 + r_a) * lda + bk0 + 64, 0, 1);
#pragma unroll
    for (int ks = 0; ks < 64; ks += 32) {
      // A fragment: lane half 0 -> K 0-7/16-23, half 1 -> K 8-15/24-31
      FragBF fa;
      const __bf16* ap = &As[(wave * 16 + l16) * 64 + ks];
      fa.f4[0] = *(const f32x4*)(ap + hf * 8);
      fa.f4[1] = *(const f32x4*)(ap + 16 + hf * 8);
#pragma unroll
      for (int nt = 0; nt < 4; ++nt) {
        // B fragment: lane half 0 -> K 0-7/16-23, half 1 -> K 8-15/24-31
        FragBF fb;
        const __bf16* bp = &Bs[(nt * 16 + l16) * 64 + ks + hf * 8];
        fb.f4[0] = *(const f32x4*)(bp);
        fb.f4[1] = *(const f32x4*)(bp + 16);
        acc[nt] = __builtin_amdgcn_wmma_f32_16x16x32_bf16(
            false, fa.v, false, fb.v, (short)0, acc[nt], false, false);
      }
    }
  }
  // epilogue: C frag layout -> row = vgpr + 8*half, col = lane%16
#pragma unroll
  for (int nt = 0; nt < 4; ++nt) {
    int col = bn0 + nt * 16 + l16;
    float bv2 = bias ? bias[col] : 0.f;
#pragma unroll
    for (int r = 0; r < 8; ++r) {
      int row = bm0 + wave * 16 + hf * 8 + r;
      float v = acc[nt][r] + bv2;
      if (act == 1) v = fmaxf(v, 0.f);
      C[(size_t)row * ldc + col] = v;
    }
  }
}

// ---------------------------------------------------------------------------
// Sequential GRU scan, one block per batch, 384 threads (one per hidden unit)
// gi = x@Wih + bih (precomputed). out gets h at [.., off : off+384].
// ---------------------------------------------------------------------------
__global__ __launch_bounds__(384)
void k_gru(const float* __restrict__ gi, const float* __restrict__ Whh,
           const float* __restrict__ bhh, float* __restrict__ out,
           int reverse, int off)
{
  __shared__ float hsh[H2_];
  int b = blockIdx.x;
  int j = threadIdx.x;
  float b_r = bhh[j], b_z = bhh[H2_ + j], b_n = bhh[2 * H2_ + j];
  hsh[j] = 0.f;
  __syncthreads();
  for (int t = 0; t < N_; ++t) {
    int tt = reverse ? (N_ - 1 - t) : t;
    float gr = b_r, gz = b_z, gn = b_n;
    for (int i = 0; i < H2_; ++i) {
      float hv = hsh[i];
      const float* wr = &Whh[(size_t)i * 3 * H2_];
      gr += hv * wr[j];
      gz += hv * wr[H2_ + j];
      gn += hv * wr[2 * H2_ + j];
    }
    const float* gir = &gi[((size_t)(b * N_ + tt)) * 3 * H2_];
    float r = sigmoidf_(gir[j] + gr);
    float z = sigmoidf_(gir[H2_ + j] + gz);
    float n = tanhf(gir[2 * H2_ + j] + r * gn);
    float hnew = (1.f - z) * n + z * hsh[j];
    __syncthreads();
    hsh[j] = hnew;
    out[((size_t)(b * N_ + tt)) * H_ + off + j] = hnew;
    __syncthreads();
  }
}

// ---------------------------------------------------------------------------
// Fused attention: scores (qk + q*sk + mask-bias) -> softmax -> out (wv + w*sv)
// One block of 96 threads per (b, head, q-row).
// ---------------------------------------------------------------------------
__global__ __launch_bounds__(96)
void k_attn(const float* __restrict__ q, const float* __restrict__ k,
            const float* __restrict__ v, const float* __restrict__ sk,
            const float* __restrict__ sv, const int* __restrict__ edu_nums,
            float* __restrict__ ao)
{
  int blk = blockIdx.x;
  int qi = blk % N_;
  int h  = (blk / N_) % HEADS_;
  int b  = blk / (N_ * HEADS_);
  __shared__ float w[N_];
  __shared__ float qv[D_];
  __shared__ float mx, sm;
  int t = threadIdx.x;
  if (t < D_) qv[t] = q[((size_t)(b * N_ + qi)) * H_ + h * D_ + t] * 0.125f;
  __syncthreads();
  int nvalid = edu_nums[b] + 1;
  {
    const float* kr  = &k[((size_t)(b * N_ + t)) * H_ + h * D_];
    const float* skr = &sk[((size_t)((b * N_ + qi) * N_ + t)) * D_];
    float s = (t < nvalid) ? 0.f : -1.0e9f;
    for (int d = 0; d < D_; ++d) s += qv[d] * (kr[d] + skr[d]);
    w[t] = s;
  }
  __syncthreads();
  if (t == 0) { float m = -3.0e38f; for (int i = 0; i < N_; ++i) m = fmaxf(m, w[i]); mx = m; }
  __syncthreads();
  float e = __expf(w[t] - mx);
  w[t] = e;
  __syncthreads();
  if (t == 0) { float s = 0.f; for (int i = 0; i < N_; ++i) s += w[i]; sm = s; }
  __syncthreads();
  float inv = 1.f / sm;
  if (t < D_) {
    const float* vbp = &v[((size_t)(b * N_)) * H_ + h * D_ + t];
    const float* svb = &sv[((size_t)((b * N_ + qi) * N_)) * D_ + t];
    float a = 0.f;
    for (int kk = 0; kk < N_; ++kk) a += w[kk] * (vbp[(size_t)kk * H_] + svb[(size_t)kk * D_]);
    ao[((size_t)(b * N_ + qi)) * H_ + h * D_ + t] = a * inv;
  }
}

// LayerNorm over rows of (x + y) (y nullable), one 128-thread block per row.
__global__ __launch_bounds__(128)
void k_ln2(const float* __restrict__ x, const float* __restrict__ y,
           const float* __restrict__ g, const float* __restrict__ bt,
           float* __restrict__ out, int cols)
{
  int row = blockIdx.x, t = threadIdx.x;
  const float* xr = x + (size_t)row * cols;
  const float* yr = y ? (y + (size_t)row * cols) : nullptr;
  float s = 0.f, s2 = 0.f;
  for (int c = t; c < cols; c += 128) {
    float v = xr[c] + (yr ? yr[c] : 0.f);
    s += v; s2 += v * v;
  }
  __shared__ float sa[128], sb[128];
  sa[t] = s; sb[t] = s2; __syncthreads();
  for (int o = 64; o > 0; o >>= 1) {
    if (t < o) { sa[t] += sa[t + o]; sb[t] += sb[t + o]; }
    __syncthreads();
  }
  float mean = sa[0] / cols;
  float var  = sb[0] / cols - mean * mean;
  float rstd = rsqrtf(var + 1e-5f);
  float* outr = out + (size_t)row * cols;
  for (int c = t; c < cols; c += 128) {
    float v = xr[c] + (yr ? yr[c] : 0.f);
    outr[c] = (v - mean) * rstd * g[c] + bt[c];
  }
}

__global__ void k_build_input(const float* __restrict__ sent, float* __restrict__ out, int n)
{
  int idx = blockIdx.x * blockDim.x + threadIdx.x; if (idx >= n) return;
  int c = idx % H_;
  int t = (idx / H_) % N_;
  int b = idx / (H_ * N_);
  out[idx] = (t == 0) ? 0.f : sent[((size_t)(b * (N_ - 1) + (t - 1))) * H_ + c];
}

__global__ void k_const_path(const int* __restrict__ speakers, const int* __restrict__ turns,
                             const float* __restrict__ emb_sp, const float* __restrict__ emb_tn,
                             const float* __restrict__ emb_pos, float* __restrict__ out, int n)
{
  int idx = blockIdx.x * blockDim.x + threadIdx.x; if (idx >= n) return;
  int rc  = idx & (P_ - 1);
  int row = idx >> 7;
  int j = row % N_;
  int i = (row / N_) % N_;
  float v;
  if (rc < 32)       v = emb_sp[speakers[row] * 32 + rc];
  else if (rc < 64)  v = emb_tn[turns[row] * 32 + (rc - 32)];
  else {
    int rel = j - i;
    rel = (rel < -11) ? -11 : (rel > 11 ? 11 : rel);
    v = emb_pos[(rel + 11) * 64 + (rc - 64)];
  }
  out[idx] = v;
}

// struct_sym[b,i,j] = (j>i) ? struct[b,j,i] : struct[b,i,j]
__global__ void k_sym(const float* __restrict__ src, float* __restrict__ dst, int n)
{
  int idx = blockIdx.x * blockDim.x + threadIdx.x; if (idx >= n) return;
  int c = idx & (P_ - 1);
  int row = idx >> 7;
  int j = row % N_;
  int i = (row / N_) % N_;
  int b = row / (N_ * N_);
  size_t sidx = (j > i) ? ((((size_t)(b * N_ + j)) * N_ + i) * P_ + c) : (size_t)idx;
  dst[idx] = src[sidx];
}

// GRU-style path update combine. "a" terms multiply nodes[b,j], "b" terms nodes[b,i].
__global__ void k_path_combine(
    const float* __restrict__ tra, const float* __restrict__ trb,
    const float* __restrict__ tza, const float* __restrict__ tzb,
    const float* __restrict__ tca, const float* __restrict__ tcb,
    const float* __restrict__ hr, const float* __restrict__ hz, const float* __restrict__ hu,
    const float* __restrict__ hx, float* __restrict__ memo, int row0, int n)
{
  int idx = blockIdx.x * blockDim.x + threadIdx.x; if (idx >= n) return;
  int c = idx & (P_ - 1);
  int r = idx >> 7;
  int row = row0 + r;
  int j = row % N_;
  int i = (row / N_) % N_;
  int b = row / (N_ * N_);
  size_t ja = ((size_t)(b * N_ + j)) * P_ + c;
  size_t ib = ((size_t)(b * N_ + i)) * P_ + c;
  float rr = sigmoidf_(tra[ja] + trb[ib] + hr[idx]);
  float zz = sigmoidf_(tza[ja] + tzb[ib] + hz[idx]);
  float uu = tanhf(tca[ja] + tcb[ib] + rr * hu[idx]);
  float hv = hx[idx];
  float nw = zz * hv + (1.f - zz) * uu;
  memo[idx] = (i > j) ? nw : hv;
}

// hidden = tanh(hidden + tA[b,j] + tB[b,i])
__global__ void k_class_elem(const float* __restrict__ tA, const float* __restrict__ tB,
                             float* __restrict__ hid, int row0, int n)
{
  int idx = blockIdx.x * blockDim.x + threadIdx.x; if (idx >= n) return;
  int c = idx % H_;
  int r = idx / H_;
  int row = row0 + r;
  int j = row % N_;
  int i = (row / N_) % N_;
  int b = row / (N_ * N_);
  hid[idx] = tanhf(hid[idx] + tA[((size_t)(b * N_ + j)) * H_ + c]
                            + tB[((size_t)(b * N_ + i)) * H_ + c]);
}

// out[row, c] = hidden[row] . W2[:, c] + b2[c]; one wave per row.
__global__ __launch_bounds__(256)
void k_w2(const float* __restrict__ hid, const float* __restrict__ W2,
          const float* __restrict__ b2, float* __restrict__ out,
          int row0, int C, int nrows)
{
  int wv = threadIdx.x >> 5, lane = threadIdx.x & 31;
  int r = blockIdx.x * 8 + wv;
  if (r >= nrows) return;
  const float* hrow = hid + (size_t)r * H_;
  float regs[24];
  for (int i = 0; i < 24; ++i) regs[i] = hrow[lane + i * 32];
  int row = row0 + r;
  for (int c = 0; c < C; ++c) {
    float a = 0.f;
    for (int i = 0; i < 24; ++i) a += regs[i] * W2[(size_t)(lane + i * 32) * C + c];
    for (int o = 16; o > 0; o >>= 1) a += __shfl_down(a, o, 32);
    if (lane == 0) out[(size_t)row * C + c] = a + b2[c];
  }
}

__global__ void k_concat(const float* __restrict__ a, const float* __restrict__ b,
                         float* __restrict__ out, int n)
{
  int idx = blockIdx.x * blockDim.x + threadIdx.x; if (idx >= n) return;
  int c = idx % (2 * H_);
  int row = idx / (2 * H_);
  out[idx] = (c < H_) ? a[(size_t)row * H_ + c] : b[(size_t)row * H_ + c - H_];
}

// ---------------------------------------------------------------------------
extern "C" void kernel_launch(void* const* d_in, const int* in_sizes, int n_in,
                              void* d_out, int out_size, void* d_ws, size_t ws_size,
                              hipStream_t stream)
{
  (void)in_sizes; (void)n_in; (void)out_size; (void)ws_size;
  const float* sentences = (const float*)d_in[0];
  const int*   speakers  = (const int*)d_in[1];
  const int*   turns     = (const int*)d_in[2];
  const int*   edu       = (const int*)d_in[3];
  int ip = 4;
  const float *Wih_f = (const float*)d_in[ip++], *Whh_f = (const float*)d_in[ip++],
              *bih_f = (const float*)d_in[ip++], *bhh_f = (const float*)d_in[ip++];
  const float *Wih_b = (const float*)d_in[ip++], *Whh_b = (const float*)d_in[ip++],
              *bih_b = (const float*)d_in[ip++], *bhh_b = (const float*)d_in[ip++];
  const float *emb_sp = (const float*)d_in[ip++], *emb_tn = (const float*)d_in[ip++],
              *emb_pos = (const float*)d_in[ip++];
  const float *Wr = (const float*)d_in[ip++], *br = (const float*)d_in[ip++],
              *Wz = (const float*)d_in[ip++], *bz = (const float*)d_in[ip++],
              *Wc = (const float*)d_in[ip++], *bc = (const float*)d_in[ip++],
              *Wu = (const float*)d_in[ip++], *bu = (const float*)d_in[ip++];
  const float *Wq = (const float*)d_in[ip++], *bq = (const float*)d_in[ip++],
              *Wk = (const float*)d_in[ip++], *bk = (const float*)d_in[ip++],
              *Wv = (const float*)d_in[ip++], *bv = (const float*)d_in[ip++],
              *Wo = (const float*)d_in[ip++], *bo = (const float*)d_in[ip++],
              *Wsk = (const float*)d_in[ip++], *bsk = (const float*)d_in[ip++],
              *Wsv = (const float*)d_in[ip++], *bsv = (const float*)d_in[ip++],
              *ln_g = (const float*)d_in[ip++], *ln_b = (const float*)d_in[ip++],
              *pn_g = (const float*)d_in[ip++], *pn_b = (const float*)d_in[ip++];
  const float *W1l = (const float*)d_in[ip++], *b1l = (const float*)d_in[ip++],
              *W2l = (const float*)d_in[ip++], *b2l = (const float*)d_in[ip++];
  const float *W1b = (const float*)d_in[ip++], *b1b = (const float*)d_in[ip++],
              *W2b = (const float*)d_in[ip++], *b2b = (const float*)d_in[ip++];

  const size_t BN  = (size_t)B_ * N_;        // 384
  const size_t BNN = (size_t)B_ * N_ * N_;   // 36864
  const int NCHUNK = 8;
  const int CH = (int)(BNN / NCHUNK);        // 4608

  float* wp = (float*)d_ws;
  auto take = [&](size_t nfloats) { float* r = wp; wp += nfloats; return r; };
  float* din    = take(BN * H_);
  float* dgo    = take(BN * H_);            // dialog_output
  float* gif    = take(BN * 3 * H2_);
  float* gib    = take(BN * 3 * H2_);
  float* nodesA = take(BN * H_);
  float* nodesB = take(BN * H_);
  float* cpath  = take(BNN * P_);
  float* ssym   = take(BNN * P_);
  float* qb = take(BN * H_); float* kb = take(BN * H_); float* vb = take(BN * H_);
  float* skb = take(BNN * D_); float* svb = take(BNN * D_);
  float* aob = take(BN * H_); float* ao2 = take(BN * H_);
  float* tra = take(BN * P_); float* trb = take(BN * P_);
  float* tza = take(BN * P_); float* tzb = take(BN * P_);
  float* tca = take(BN * P_); float* tcb = take(BN * P_);
  float* pnc = take((size_t)CH * P_);
  float* hrc = take((size_t)CH * P_); float* hzc = take((size_t)CH * P_);
  float* huc = take((size_t)CH * P_);
  float* ma  = take(BN * 2 * H_);
  float* tAl = take(BN * H_); float* tBl = take(BN * H_);
  float* tAb = take(BN * H_); float* tBb = take(BN * H_);
  float* hid = take((size_t)CH * H_);

  float* outp    = (float*)d_out;
  float* link_p  = outp;
  float* label_p = outp + BNN;
  float* mem_p   = outp + BNN + BNN * REL_;

  auto gemm = [&](const float* A, const float* Bm, const float* bias, float* C,
                  int M, int Nn, int K, int lda, int ldb, int ldc, int act) {
    dim3 g((Nn + 63) / 64, (M + 63) / 64);
    k_gemm<<<g, dim3(128), 0, stream>>>(A, Bm, bias, C, M, Nn, K, lda, ldb, ldc, act);
  };

  // ---- dialog input + BiGRU ----
  k_build_input<<<(int)((BN * H_ + 255) / 256), 256, 0, stream>>>(sentences, din, (int)(BN * H_));
  gemm(din, Wih_f, bih_f, gif, (int)BN, 3 * H2_, H_, H_, 3 * H2_, 3 * H2_, 0);
  gemm(din, Wih_b, bih_b, gib, (int)BN, 3 * H2_, H_, H_, 3 * H2_, 3 * H2_, 0);
  k_gru<<<B_, 384, 0, stream>>>(gif, Whh_f, bhh_f, dgo, 0, 0);
  k_gru<<<B_, 384, 0, stream>>>(gib, Whh_b, bhh_b, dgo, 1, H2_);

  // ---- const path ----
  k_const_path<<<(int)((BNN * P_ + 255) / 256), 256, 0, stream>>>(
      speakers, turns, emb_sp, emb_tn, emb_pos, cpath, (int)(BNN * P_));

  // ---- layers ----
  const float* nodes_in = dgo;
  float* nodes_out = nodesA;
  for (int l = 0; l < LAYERS_; ++l) {
    float* mem_l = mem_p + (size_t)l * BNN * P_;
    if (l == 0) {
      hipMemsetAsync(ssym, 0, BNN * P_ * sizeof(float), stream);
    } else {
      const float* prev = mem_p + (size_t)(l - 1) * BNN * P_;
      k_sym<<<(int)((BNN * P_ + 255) / 256), 256, 0, stream>>>(prev, ssym, (int)(BNN * P_));
    }
    // path LN + sk/sv projections (chunked)
    for (int ch = 0; ch < NCHUNK; ++ch) {
      size_t off = (size_t)ch * CH;
      k_ln2<<<CH, 128, 0, stream>>>(ssym + off * P_, cpath + off * P_, pn_g, pn_b, pnc, P_);
      gemm(pnc, Wsk, bsk, skb + off * D_, CH, D_, P_, P_, D_, D_, 0);
      gemm(pnc, Wsv, bsv, svb + off * D_, CH, D_, P_, P_, D_, D_, 0);
    }
    // q,k,v
    gemm(nodes_in, Wq, bq, qb, (int)BN, H_, H_, H_, H_, H_, 0);
    gemm(nodes_in, Wk, bk, kb, (int)BN, H_, H_, H_, H_, H_, 0);
    gemm(nodes_in, Wv, bv, vb, (int)BN, H_, H_, H_, H_, H_, 0);
    // fused attention
    k_attn<<<B_ * HEADS_ * N_, 96, 0, stream>>>(qb, kb, vb, skb, svb, edu, aob);
    // output projection + relu, then residual LN
    gemm(aob, Wo, bo, ao2, (int)BN, H_, H_, H_, H_, H_, 1);
    k_ln2<<<(int)BN, 128, 0, stream>>>(nodes_in, ao2, ln_g, ln_b, nodes_out, H_);
    // path update: factored node terms ("a" = rows 0..767 -> nodes[b,j])
    gemm(nodes_out, Wr,                     nullptr, tra, (int)BN, P_, H_, H_, P_, P_, 0);
    gemm(nodes_out, Wr + (size_t)768 * P_,  nullptr, trb, (int)BN, P_, H_, H_, P_, P_, 0);
    gemm(nodes_out, Wz,                     nullptr, tza, (int)BN, P_, H_, H_, P_, P_, 0);
    gemm(nodes_out, Wz + (size_t)768 * P_,  nullptr, tzb, (int)BN, P_, H_, H_, P_, P_, 0);
    gemm(nodes_out, Wc,                     bc,      tca, (int)BN, P_, H_, H_, P_, P_, 0);
    gemm(nodes_out, Wc + (size_t)768 * P_,  nullptr, tcb, (int)BN, P_, H_, H_, P_, P_, 0);
    // hx-dependent terms + combine (chunked)
    for (int ch = 0; ch < NCHUNK; ++ch) {
      size_t off = (size_t)ch * CH;
      gemm(ssym + off * P_, Wr + (size_t)1536 * P_, br, hrc, CH, P_, P_, P_, P_, P_, 0);
      gemm(ssym + off * P_, Wz + (size_t)1536 * P_, bz, hzc, CH, P_, P_, P_, P_, P_, 0);
      gemm(ssym + off * P_, Wu,                     bu, huc, CH, P_, P_, P_, P_, P_, 0);
      k_path_combine<<<(CH * P_ + 255) / 256, 256, 0, stream>>>(
          tra, trb, tza, tzb, tca, tcb, hrc, hzc, huc,
          ssym + off * P_, mem_l + off * P_, (int)off, CH * P_);
    }
    nodes_in = nodes_out;
    nodes_out = (nodes_out == nodesA) ? nodesB : nodesA;
  }

  // ---- classifiers (factored 3200-K GEMM: node halves + struct tail) ----
  k_concat<<<(int)((BN * 2 * H_ + 255) / 256), 256, 0, stream>>>(din, dgo, ma, (int)(BN * 2 * H_));
  gemm(ma, W1l,                       b1l,     tAl, (int)BN, H_, 2 * H_, 2 * H_, H_, H_, 0);
  gemm(ma, W1l + (size_t)1536 * H_,   nullptr, tBl, (int)BN, H_, 2 * H_, 2 * H_, H_, H_, 0);
  gemm(ma, W1b,                       b1b,     tAb, (int)BN, H_, 2 * H_, 2 * H_, H_, H_, 0);
  gemm(ma, W1b + (size_t)1536 * H_,   nullptr, tBb, (int)BN, H_, 2 * H_, 2 * H_, H_, H_, 0);
  const float* sp_final = mem_p + (size_t)2 * BNN * P_;
  for (int ch = 0; ch < NCHUNK; ++ch) {
    size_t off = (size_t)ch * CH;
    // link
    gemm(sp_final + off * P_, W1l + (size_t)3072 * H_, nullptr, hid, CH, H_, P_, P_, H_, H_, 0);
    k_class_elem<<<(CH * H_ + 255) / 256, 256, 0, stream>>>(tAl, tBl, hid, (int)off, CH * H_);
    k_w2<<<(CH + 7) / 8, 256, 0, stream>>>(hid, W2l, b2l, link_p, (int)off, 1, CH);
    // label
    gemm(sp_final + off * P_, W1b + (size_t)3072 * H_, nullptr, hid, CH, H_, P_, P_, H_, H_, 0);
    k_class_elem<<<(CH * H_ + 255) / 256, 256, 0, stream>>>(tAb, tBb, hid, (int)off, CH * H_);
    k_w2<<<(CH + 7) / 8, 256, 0, stream>>>(hid, W2b, b2b, label_p, (int)off, REL_, CH);
  }
}
